// ProteinMPNN_5781025980977
// MI455X (gfx1250) — matrix operand order repaired and checked by reading, not compiled, not run on hardware
//
#include <hip/hip_runtime.h>

#define BB 4
#define LL 2048
#define KK 48
#define HH 128
#define FFD 512
#define SCALE_INV (1.0f/30.0f)

typedef __attribute__((ext_vector_type(16))) __bf16 bf16x16;
typedef __attribute__((ext_vector_type(8)))  float f32x8;
typedef __attribute__((ext_vector_type(8)))  unsigned int u32x8;

// ---------- helpers ----------
// pack two f32 -> two bf16 (round-to-nearest) in one dword: rounding adds + v_perm_b32
__device__ __forceinline__ unsigned int pk2bf(float a, float b){
  unsigned int ua=__float_as_uint(a) + 0x8000u;
  unsigned int ub=__float_as_uint(b) + 0x8000u;
#if __has_builtin(__builtin_amdgcn_perm)
  return __builtin_amdgcn_perm(ub, ua, 0x07060302u); // D = [b3 b2 a3 a2]
#else
  return (ua>>16) | (ub & 0xFFFF0000u);
#endif
}
__device__ __forceinline__ unsigned short bf16b(float a){
  return (unsigned short)((__float_as_uint(a) + 0x8000u)>>16);
}
// branchless tanh-GELU in sigmoid form: x * sigmoid(1.702-ish cubic arg)
// gelu(x) ~= x / (1 + exp2(K * x * (1 + 0.044715 x^2))), K = -2*0.7978845608*log2(e)
__device__ __forceinline__ float gelu_f(float x){
  float x2 = x*x;
  float t  = __builtin_fmaf(x2, 0.044715f, 1.0f);
  float arg = (-2.302206815f * x) * t;
  float e = __builtin_amdgcn_exp2f(arg);
#if __has_builtin(__builtin_amdgcn_rcpf)
  return x * __builtin_amdgcn_rcpf(1.0f + e);
#else
  return x / (1.0f + e);
#endif
}
__device__ __forceinline__ f32x8 wmma_bf16(u32x8 a, u32x8 b, f32x8 c){
  return __builtin_amdgcn_wmma_f32_16x16x32_bf16(false, __builtin_bit_cast(bf16x16,a),
                                                 false, __builtin_bit_cast(bf16x16,b),
                                                 (short)0, c, false, false);
}
// B fragment: pre-packed bf16 tile (32x16), 8 dwords per lane, contiguous.
__device__ __forceinline__ u32x8 load_bfrag(const unsigned int* pk, int tile, int lane){
  const uint4* p = reinterpret_cast<const uint4*>(pk) + (((size_t)tile<<5) + (size_t)lane)*2;
  uint4 x=p[0], y=p[1];
  u32x8 r={x.x,x.y,x.z,x.w,y.x,y.y,y.z,y.w};
  return r;
}
// A fragment (16x32 bf16) loaded straight from a global fp32 row, converted inline.
__device__ __forceinline__ u32x8 load_afrag_g(const float* rp, int col0){
  float4 f0=*reinterpret_cast<const float4*>(rp+col0);
  float4 f1=*reinterpret_cast<const float4*>(rp+col0+4);
  float4 f2=*reinterpret_cast<const float4*>(rp+col0+16);
  float4 f3=*reinterpret_cast<const float4*>(rp+col0+20);
  u32x8 r={pk2bf(f0.x,f0.y),pk2bf(f0.z,f0.w),pk2bf(f1.x,f1.y),pk2bf(f1.z,f1.w),
           pk2bf(f2.x,f2.y),pk2bf(f2.z,f2.w),pk2bf(f3.x,f3.y),pk2bf(f3.z,f3.w)};
  return r;
}
// A fragment from a row-major bf16 LDS tile.
__device__ __forceinline__ u32x8 load_afrag_l(const unsigned short* t, int rowlen, int m, int col0){
  const uint4* p0=reinterpret_cast<const uint4*>(t + m*rowlen + col0);
  const uint4* p1=reinterpret_cast<const uint4*>(t + m*rowlen + col0 + 16);
  uint4 x=*p0,y=*p1;
  u32x8 r={x.x,x.y,x.z,x.w,y.x,y.y,y.z,y.w};
  return r;
}
// C/D tile (8 f32 per lane) -> row-major bf16 LDS tile
__device__ __forceinline__ void store_ctile(unsigned short* t, int rowlen, int nt, int lane, f32x8 acc){
  int n = nt*16 + (lane&15);
  int mb = (lane>>4)*8;
  #pragma unroll
  for(int v=0;v<8;v++) t[(mb+v)*rowlen + n] = bf16b(acc[v]);
}

// ---------- weight pack: fp32 [Kdim x N] row-major -> bf16 B-fragment order ----------
__global__ void k_pack(const float* __restrict__ W, unsigned int* __restrict__ pk, int Kdim, int N){
  int i = blockIdx.x*blockDim.x + threadIdx.x;
  int total = (Kdim>>5)*(N>>4)*256;     // 256 dwords per 32x16 tile
  if(i>=total) return;
  int NT = N>>4;
  int tile = i>>8; int rem=i&255; int lane=rem>>3; int j=rem&7;
  int kc = tile/NT, nt = tile - kc*NT;
  int n = lane&15;
  int k = (kc<<5) + ((lane>>4)<<3) + ((j&3)<<1) + ((j>>2)<<4);
  float a = W[(size_t)k*N + nt*16 + n];
  float b = W[(size_t)(k+1)*N + nt*16 + n];
  pk[i] = pk2bf(a,b);
}

// ---------- kernel 1: node message MLP + masked sum over K ----------
__global__ __launch_bounds__(96) void k_node_msg(
    const float* __restrict__ hV, const float* __restrict__ hE,
    const float* __restrict__ maskA, const int* __restrict__ Eidx,
    const unsigned int* __restrict__ pkW1, const unsigned int* __restrict__ pkW2,
    const unsigned int* __restrict__ pkW3,
    const float* __restrict__ b1, const float* __restrict__ b2, const float* __restrict__ b3,
    float* __restrict__ dh_out)
{
  __shared__ __align__(16) unsigned char smem[3*8192 + 3*128*4];
  const int tid=threadIdx.x, wave=tid>>5, lane=tid&31;
  const int m=lane&15, hi=lane>>4;
  const int blk=blockIdx.x;
  const int b=blk>>11, l=blk&(LL-1);
  const int node=b*LL+l;
  const int k0=wave*16;
  unsigned short* t0=(unsigned short*)(smem + wave*8192);
  unsigned short* t1=t0 + 2048;
  float* dhp=(float*)(smem + 3*8192);

  const float* rp_v = hV + (size_t)node*HH;
  const int erow = node*KK + k0 + m;
  const float* rp_e = hE + (size_t)erow*HH;
  const int nidx = Eidx[erow];
  const float* rp_g = hV + ((size_t)b*LL + nidx)*HH;

  // ---- layer 1: [16x384] @ W1 ----
  f32x8 acc[8];
  #pragma unroll
  for(int nt=0;nt<8;nt++){ f32x8 z={0,0,0,0,0,0,0,0}; acc[nt]=z; }
  for(int c=0;c<12;c++){
    const float* rp = (c<4)? rp_v : ((c<8)? rp_e : rp_g);
    int col0 = ((c&3)<<5) + hi*8;
    u32x8 a = load_afrag_g(rp, col0);
    #pragma unroll
    for(int nt=0;nt<8;nt++){
      u32x8 bf = load_bfrag(pkW1, c*8+nt, lane);
      acc[nt] = wmma_bf16(a, bf, acc[nt]);
    }
  }
  #pragma unroll
  for(int nt=0;nt<8;nt++){
    float bb=b1[nt*16+m];
    f32x8 v=acc[nt];
    #pragma unroll
    for(int j=0;j<8;j++) v[j]=gelu_f(v[j]+bb);
    store_ctile(t0,HH,nt,lane,v);
  }
  // ---- layer 2 ----
  for(int nt=0;nt<8;nt++){
    f32x8 a2={0,0,0,0,0,0,0,0};
    for(int c=0;c<4;c++){
      u32x8 af=load_afrag_l(t0,HH,m,(c<<5)+hi*8);
      u32x8 bf=load_bfrag(pkW2,c*8+nt,lane);
      a2=wmma_bf16(af,bf,a2);
    }
    float bb=b2[nt*16+m];
    #pragma unroll
    for(int j=0;j<8;j++) a2[j]=gelu_f(a2[j]+bb);
    store_ctile(t1,HH,nt,lane,a2);
  }
  // ---- layer 3 + mask + K-reduction ----
  float mk[8];
  const int mb=hi*8;
  #pragma unroll
  for(int v=0;v<8;v++) mk[v]=maskA[(size_t)node*KK + k0 + mb + v];
  for(int nt=0;nt<8;nt++){
    f32x8 a3={0,0,0,0,0,0,0,0};
    for(int c=0;c<4;c++){
      u32x8 af=load_afrag_l(t1,HH,m,(c<<5)+hi*8);
      u32x8 bf=load_bfrag(pkW3,c*8+nt,lane);
      a3=wmma_bf16(af,bf,a3);
    }
    float bb=b3[nt*16+m];
    float s=0.f;
    #pragma unroll
    for(int v=0;v<8;v++) s += (a3[v]+bb)*mk[v];
    s += __shfl_xor(s,16,32);
    if(hi==0) dhp[wave*128 + nt*16 + m] = s;   // per-wave partial (deterministic)
  }
  __syncthreads();
  for(int i=tid;i<128;i+=96)
    dh_out[(size_t)node*HH + i] = dhp[i] + dhp[128+i] + dhp[256+i];
}

// ---------- kernel 2: residual + LN1 + FFN (WMMA) + LN2 + mask ----------
__global__ __launch_bounds__(64) void k_ffn(
    const float* __restrict__ hV, const float* __restrict__ dh,
    const float* __restrict__ maskV,
    const unsigned int* __restrict__ pkWin, const unsigned int* __restrict__ pkWout,
    const float* __restrict__ bIn, const float* __restrict__ bOut,
    const float* __restrict__ g1, const float* __restrict__ be1,
    const float* __restrict__ g2, const float* __restrict__ be2,
    float* __restrict__ hV_out)
{
  __shared__ __align__(16) unsigned char smem[2*28800];
  const int tid=threadIdx.x, wave=tid>>5, lane=tid&31;
  const int m=lane&15, hi=lane>>4;
  unsigned char* wb = smem + wave*28800;
  float* z = (float*)wb;                          // 16x128 f32
  unsigned short* xt = (unsigned short*)(wb+8192); // 16x128 bf16
  unsigned short* H1 = (unsigned short*)(wb+12288);// 16x512 bf16
  float* mrow=(float*)(wb+28672);
  float* rrow=mrow+16;
  const int rowbase = blockIdx.x*32 + wave*16;

  // residual
  for(int it=0;it<64;it++){
    int idx=it*32+lane; int r=idx>>7, c=idx&127;
    size_t gr=(size_t)(rowbase+r)*HH+c;
    z[idx]=hV[gr]+dh[gr]*SCALE_INV;
  }
  // LN1 stats
  {
    int r=lane>>1, half=lane&1;
    float s=0.f,ss=0.f;
    for(int c=0;c<64;c++){float x=z[r*128+half*64+c]; s+=x; ss+=x*x;}
    s+=__shfl_xor(s,1,32); ss+=__shfl_xor(ss,1,32);
    float mean=s*(1.f/128.f), var=ss*(1.f/128.f)-mean*mean;
    if(half==0){mrow[r]=mean; rrow[r]=rsqrtf(var+1e-5f);}
  }
  // y = LN1(x); keep f32 in z, bf16 in xt
  for(int it=0;it<64;it++){
    int idx=it*32+lane; int r=idx>>7, c=idx&127;
    float y=(z[idx]-mrow[r])*rrow[r]*g1[c]+be1[c];
    z[idx]=y; xt[idx]=bf16b(y);
  }
  // FFN: H1 = gelu(y @ Win + bin)
  for(int t=0;t<32;t++){
    f32x8 acc={0,0,0,0,0,0,0,0};
    for(int c=0;c<4;c++){
      u32x8 af=load_afrag_l(xt,HH,m,(c<<5)+hi*8);
      u32x8 bf=load_bfrag(pkWin,c*32+t,lane);
      acc=wmma_bf16(af,bf,acc);
    }
    float bb=bIn[t*16+m];
    #pragma unroll
    for(int v=0;v<8;v++) acc[v]=gelu_f(acc[v]+bb);
    store_ctile(H1,FFD,t,lane,acc);
  }
  // z += H1 @ Wout + bout
  for(int nt=0;nt<8;nt++){
    f32x8 acc={0,0,0,0,0,0,0,0};
    for(int c=0;c<16;c++){
      u32x8 af=load_afrag_l(H1,FFD,m,(c<<5)+hi*8);
      u32x8 bf=load_bfrag(pkWout,c*8+nt,lane);
      acc=wmma_bf16(af,bf,acc);
    }
    float bb=bOut[nt*16+m];
    int n=nt*16+m, mb=hi*8;
    #pragma unroll
    for(int v=0;v<8;v++){ int idx=(mb+v)*128+n; z[idx]=z[idx]+acc[v]+bb; }
  }
  // LN2 stats
  {
    int r=lane>>1, half=lane&1;
    float s=0.f,ss=0.f;
    for(int c=0;c<64;c++){float x=z[r*128+half*64+c]; s+=x; ss+=x*x;}
    s+=__shfl_xor(s,1,32); ss+=__shfl_xor(ss,1,32);
    float mean=s*(1.f/128.f), var=ss*(1.f/128.f)-mean*mean;
    if(half==0){mrow[r]=mean; rrow[r]=rsqrtf(var+1e-5f);}
  }
  // LN2 + mask, contiguous store
  for(int it=0;it<64;it++){
    int idx=it*32+lane; int r=idx>>7, c=idx&127;
    size_t row=(size_t)(rowbase+r);
    float val=(z[idx]-mrow[r])*rrow[r]*g2[c]+be2[c];
    hV_out[row*HH+c]=val*maskV[row];
  }
}

// ---------- kernel 3: edge MLP + residual + LN3 ----------
__global__ __launch_bounds__(96) void k_edge(
    const float* __restrict__ hVn, const float* __restrict__ hE,
    const unsigned int* __restrict__ pkW11, const unsigned int* __restrict__ pkW12,
    const unsigned int* __restrict__ pkW13,
    const float* __restrict__ b11, const float* __restrict__ b12, const float* __restrict__ b13,
    const int* __restrict__ Eidx,
    const float* __restrict__ g3, const float* __restrict__ be3,
    float* __restrict__ hE_out)
{
  __shared__ __align__(16) unsigned char smem[3*16512];
  const int tid=threadIdx.x, wave=tid>>5, lane=tid&31;
  const int m=lane&15, hi=lane>>4, mb=hi*8;
  const int blk=blockIdx.x;
  const int b=blk>>11, l=blk&(LL-1);
  const int node=b*LL+l;
  const int k0=wave*16;
  unsigned char* wb = smem + wave*16512;
  unsigned short* t0=(unsigned short*)wb;
  unsigned short* t1=(unsigned short*)(wb+4096);
  float* S=(float*)(wb+8192);
  float* mrow=(float*)(wb+16384);
  float* rrow=mrow+16;

  const float* rp_v = hVn + (size_t)node*HH;
  const int erow = node*KK + k0 + m;
  const float* rp_e = hE + (size_t)erow*HH;
  const int nidx = Eidx[erow];
  const float* rp_g = hVn + ((size_t)b*LL + nidx)*HH;

  // layer 1
  f32x8 acc[8];
  #pragma unroll
  for(int nt=0;nt<8;nt++){ f32x8 zz={0,0,0,0,0,0,0,0}; acc[nt]=zz; }
  for(int c=0;c<12;c++){
    const float* rp = (c<4)? rp_v : ((c<8)? rp_e : rp_g);
    int col0 = ((c&3)<<5) + hi*8;
    u32x8 a = load_afrag_g(rp, col0);
    #pragma unroll
    for(int nt=0;nt<8;nt++){
      u32x8 bf = load_bfrag(pkW11, c*8+nt, lane);
      acc[nt] = wmma_bf16(a, bf, acc[nt]);
    }
  }
  #pragma unroll
  for(int nt=0;nt<8;nt++){
    float bb=b11[nt*16+m];
    f32x8 v=acc[nt];
    #pragma unroll
    for(int j=0;j<8;j++) v[j]=gelu_f(v[j]+bb);
    store_ctile(t0,HH,nt,lane,v);
  }
  // layer 2
  for(int nt=0;nt<8;nt++){
    f32x8 a2={0,0,0,0,0,0,0,0};
    for(int c=0;c<4;c++){
      u32x8 af=load_afrag_l(t0,HH,m,(c<<5)+hi*8);
      u32x8 bf=load_bfrag(pkW12,c*8+nt,lane);
      a2=wmma_bf16(af,bf,a2);
    }
    float bb=b12[nt*16+m];
    #pragma unroll
    for(int j=0;j<8;j++) a2[j]=gelu_f(a2[j]+bb);
    store_ctile(t1,HH,nt,lane,a2);
  }
  // layer 3 + residual into S
  const size_t ebase = ((size_t)node*KK + k0)*HH;
  for(int nt=0;nt<8;nt++){
    f32x8 a3={0,0,0,0,0,0,0,0};
    for(int c=0;c<4;c++){
      u32x8 af=load_afrag_l(t1,HH,m,(c<<5)+hi*8);
      u32x8 bf=load_bfrag(pkW13,c*8+nt,lane);
      a3=wmma_bf16(af,bf,a3);
    }
    float bb=b13[nt*16+m];
    int n=nt*16+m;
    #pragma unroll
    for(int v=0;v<8;v++){
      int idx=(mb+v)*HH+n;
      S[idx]=hE[ebase+idx]+a3[v]+bb;
    }
  }
  // per-row LN3 stats
  {
    int r=lane>>1, half=lane&1;
    float s=0.f,ss=0.f;
    for(int c=0;c<64;c++){float x=S[r*128+half*64+c]; s+=x; ss+=x*x;}
    s+=__shfl_xor(s,1,32); ss+=__shfl_xor(ss,1,32);
    float mean=s*(1.f/128.f), var=ss*(1.f/128.f)-mean*mean;
    if(half==0){mrow[r]=mean; rrow[r]=rsqrtf(var+1e-5f);}
  }
  // normalize + contiguous coalesced store of the whole 16x128 tile
  for(int it=0;it<64;it++){
    int idx=it*32+lane; int r=idx>>7, c=idx&127;
    float val=(S[idx]-mrow[r])*rrow[r]*g3[c]+be3[c];
    hE_out[ebase+idx]=val;
  }
}

// ---------- host ----------
extern "C" void kernel_launch(void* const* d_in, const int* in_sizes, int n_in,
                              void* d_out, int out_size, void* d_ws, size_t ws_size,
                              hipStream_t stream) {
  const float* hV   =(const float*)d_in[0];
  const float* hE   =(const float*)d_in[1];
  const float* maskV=(const float*)d_in[2];
  const float* maskA=(const float*)d_in[3];
  const float* W1 =(const float*)d_in[4];  const float* b1 =(const float*)d_in[5];
  const float* W2 =(const float*)d_in[6];  const float* b2 =(const float*)d_in[7];
  const float* W3 =(const float*)d_in[8];  const float* b3 =(const float*)d_in[9];
  const float* W11=(const float*)d_in[10]; const float* b11=(const float*)d_in[11];
  const float* W12=(const float*)d_in[12]; const float* b12=(const float*)d_in[13];
  const float* W13=(const float*)d_in[14]; const float* b13=(const float*)d_in[15];
  const float* Win=(const float*)d_in[16]; const float* bIn=(const float*)d_in[17];
  const float* Wout=(const float*)d_in[18];const float* bOut=(const float*)d_in[19];
  const float* g1=(const float*)d_in[20];  const float* be1=(const float*)d_in[21];
  const float* g2=(const float*)d_in[22];  const float* be2=(const float*)d_in[23];
  const float* g3=(const float*)d_in[24];  const float* be3=(const float*)d_in[25];
  const int* Eidx=(const int*)d_in[26];

  unsigned char* ws=(unsigned char*)d_ws;
  unsigned int* pkW1  =(unsigned int*)(ws + 0);
  unsigned int* pkW2  =(unsigned int*)(ws + 98304);
  unsigned int* pkW3  =(unsigned int*)(ws + 131072);
  unsigned int* pkW11 =(unsigned int*)(ws + 163840);
  unsigned int* pkW12 =(unsigned int*)(ws + 262144);
  unsigned int* pkW13 =(unsigned int*)(ws + 294912);
  unsigned int* pkWin =(unsigned int*)(ws + 327680);
  unsigned int* pkWout=(unsigned int*)(ws + 458752);
  float* dh           =(float*)(ws + 589824);

  auto pack=[&](const float* W, unsigned int* pk, int K, int N){
    int total=(K/32)*(N/16)*256;
    k_pack<<<(total+255)/256,256,0,stream>>>(W,pk,K,N);
  };
  pack(W1 ,pkW1 ,384,128);
  pack(W2 ,pkW2 ,128,128);
  pack(W3 ,pkW3 ,128,128);
  pack(W11,pkW11,384,128);
  pack(W12,pkW12,128,128);
  pack(W13,pkW13,128,128);
  pack(Win,pkWin,128,512);
  pack(Wout,pkWout,512,128);

  float* hVout=(float*)d_out;
  float* hEout=hVout + (size_t)BB*LL*HH;

  k_node_msg<<<BB*LL,96,0,stream>>>(hV,hE,maskA,Eidx,pkW1,pkW2,pkW3,b1,b2,b3,dh);
  k_ffn<<<(BB*LL)/32,64,0,stream>>>(hV,dh,maskV,pkWin,pkWout,bIn,bOut,g1,be1,g2,be2,hVout);
  k_edge<<<BB*LL,96,0,stream>>>(hVout,hE,pkW11,pkW12,pkW13,b11,b12,b13,Eidx,g3,be3,hEout);
}
